// RoPEViTBlock_24111946400308
// MI455X (gfx1250) — compile-verified, matrix operand order checked by reading, not run on hardware
//
#include <hip/hip_runtime.h>
#include <math.h>

// ---------------------------------------------------------------------------
// RoPE ViT block for MI455X (gfx1250, wave32, WMMA 16x16x32 bf16 / f32 accum)
// GEMMs use the Tensor Data Mover (tensor_load_to_lds) for tile staging with
// double-buffered LDS and TENSORcnt-based software pipelining.
// ---------------------------------------------------------------------------
static constexpr int Bc = 4;
static constexpr int Sc = 2048;
static constexpr int Dc = 768;
static constexpr int Hc = 8;
static constexpr int DHc = 96;
static constexpr int HIDc = 3072;
static constexpr int Mrows = Bc * Sc;          // 8192

typedef __bf16 bf16_t;
typedef __attribute__((ext_vector_type(16))) __bf16 v16bf;
typedef __attribute__((ext_vector_type(8)))  __bf16 v8bf;
typedef __attribute__((ext_vector_type(8)))  float  v8f;
typedef __attribute__((ext_vector_type(4)))  unsigned u32x4;
typedef __attribute__((ext_vector_type(8)))  unsigned u32x8;

// ---- fp32 -> bf16 round-to-nearest-even ----
__device__ __forceinline__ bf16_t f2bf(float f) {
    union { float f; unsigned u; } in; in.f = f;
    unsigned r = in.u + 0x7fffu + ((in.u >> 16) & 1u);
    union { unsigned short s; bf16_t h; } out;
    out.s = (unsigned short)(r >> 16);
    return out.h;
}

// ---- WMMA fragment loader -------------------------------------------------
// A (16xK) and B-as-N-major (16 rows x K) both use: lane%16 selects the
// row/col index, and each lane holds two contiguous 8-half K-runs:
//   lanes 0-15 : K = [0..7]  and [16..23]   (relative to frag K base)
//   lanes 16-31: K = [8..15] and [24..31]
// With row-major (K-contiguous) storage this is exactly two b128 loads.
__device__ __forceinline__ v16bf load_frag(const bf16_t* __restrict__ base,
                                           int row_stride, int lane) {
    const int idx   = lane & 15;
    const int base0 = (lane >> 4) * 8;      // 0 or 8
    const bf16_t* p = base + idx * row_stride;
    v8bf lo = *(const v8bf*)(p + base0);
    v8bf hi = *(const v8bf*)(p + 16 + base0);
    v16bf f;
#pragma unroll
    for (int i = 0; i < 8; ++i) { f[i] = lo[i]; f[i + 8] = hi[i]; }
    return f;
}

// ---- Tensor Data Mover: async 2-D tile load global -> LDS -----------------
// Builds D# group0 (4 SGPRs) + group1 (8 SGPRs) per CDNA5 ISA 8.3/8.4 and
// issues TENSOR_LOAD_TO_LDS (tracked by TENSORcnt). 2-byte elements.
// When `pad` is set, 8 DWORDs of LDS padding are inserted every 16 DWORDs,
// so a 32-half row lands on a 48-half (96 B) stride (bank-conflict pad,
// still 16 B aligned for b128 fragment loads).
__device__ __forceinline__ unsigned lds_off(const void* p) {
    return (unsigned)(uintptr_t)p;          // flat LDS addr truncates to offset
}

__device__ __forceinline__ void tdm_load_tile_2d(
    unsigned lds_byte, const bf16_t* g, unsigned tile_x, unsigned tile_y,
    unsigned row_stride_elems, unsigned tdim0, unsigned tdim1, bool pad) {
    unsigned long long ga = (unsigned long long)(uintptr_t)g;
    u32x4 g0;
    g0[0] = 1u;                                         // count=1 (valid D#)
    g0[1] = lds_byte;                                   // lds_addr
    g0[2] = (unsigned)ga;                               // global_addr[31:0]
    g0[3] = (unsigned)((ga >> 32) & 0x1FFFFFFu)         // global_addr[56:32]
          | (2u << 30);                                 // type=2 ("image")
    u32x8 g1;
    g1[0] = (1u << 16)                                  // data_size=2 bytes
          | (pad ? ((1u << 20)                          // pad_enable
                  | (3u << 22)                          // pad_interval=16 DW
                  | (7u << 25)) : 0u);                  // pad_amount=8 DW
    g1[1] = (tdim0 & 0xFFFFu) << 16;                    // tensor_dim0[15:0]
    g1[2] = ((tdim0 >> 16) & 0xFFFFu)                   // tensor_dim0[31:16]
          | ((tdim1 & 0xFFFFu) << 16);                  // tensor_dim1[15:0]
    g1[3] = ((tdim1 >> 16) & 0xFFFFu)                   // tensor_dim1[31:16]
          | ((tile_x & 0xFFFFu) << 16);                 // tile_dim0
    g1[4] = (tile_y & 0xFFFFu);                         // tile_dim1 (dim2=0)
    g1[5] = row_stride_elems;                           // dim0_stride[31:0]
    g1[6] = 0u;                                         // stride[47:32], d1s lo
    g1[7] = 0u;
    asm volatile("tensor_load_to_lds %0, %1" :: "s"(g0), "s"(g1) : "memory");
}

// ---------------------------------------------------------------------------
// Kernel 0: fp32 W[K][N] -> bf16 W^T[N][K]
// ---------------------------------------------------------------------------
__global__ void k_convert_transpose(const float* __restrict__ W,
                                    bf16_t* __restrict__ Wt, int K, int N) {
    int i = blockIdx.x * 256 + threadIdx.x;
    if (i >= N * K) return;
    int n = i / K, k = i - n * K;
    Wt[i] = f2bf(W[(size_t)k * N + n]);
}

// ---------------------------------------------------------------------------
// Kernel 1: row LayerNorm (D=768) -> bf16
// ---------------------------------------------------------------------------
__global__ __launch_bounds__(256)
void k_layernorm_bf16(const float* __restrict__ x, const float* __restrict__ g,
                      const float* __restrict__ b, bf16_t* __restrict__ out) {
    __shared__ float s_sum[256], s_sq[256];
    const int row = blockIdx.x;
    const int t   = threadIdx.x;
    const float* xr = x + (size_t)row * Dc;
    float sum = 0.f, sq = 0.f;
#pragma unroll
    for (int i = 0; i < 3; ++i) {
        float v = xr[t + i * 256];
        sum += v; sq += v * v;
    }
    s_sum[t] = sum; s_sq[t] = sq;
    __syncthreads();
    for (int off = 128; off > 0; off >>= 1) {
        if (t < off) { s_sum[t] += s_sum[t + off]; s_sq[t] += s_sq[t + off]; }
        __syncthreads();
    }
    const float mu  = s_sum[0] * (1.f / Dc);
    const float var = s_sq[0] * (1.f / Dc) - mu * mu;
    const float inv = rsqrtf(var + 1e-6f);
    bf16_t* orow = out + (size_t)row * Dc;
#pragma unroll
    for (int i = 0; i < 3; ++i) {
        int c = t + i * 256;
        orow[c] = f2bf((xr[c] - mu) * inv * g[c] + b[c]);
    }
}

// ---------------------------------------------------------------------------
// Kernel 2: tiled bf16 GEMM  C[M,N] = A[M,K] @ W[K,N] (+ epilogue)
//   A: bf16 row-major [M][K], Wt: bf16 [N][K] (pre-transposed)
//   EPI 0: outF = acc + bias              (fp32)
//   EPI 1: outF = acc + bias + res        (fp32)
//   EPI 2: outH = gelu(acc + bias)        (bf16)
// Block 256 thr = 8 waves; tile 128x128x32; wave computes 32x64 (2x4 tiles).
// Tile staging via TDM (wave 0 issues tensor_load_to_lds, double-buffered):
//   wait(tile i) -> barrier -> issue(tile i+1) -> 8x WMMA on tile i
// ---------------------------------------------------------------------------
template <int EPI>
__global__ __launch_bounds__(256)
void k_gemm_bf16(const bf16_t* __restrict__ A, const bf16_t* __restrict__ Wt,
                 const float* __restrict__ bias, const float* __restrict__ res,
                 float* __restrict__ outF, bf16_t* __restrict__ outH,
                 int M, int N, int K) {
    constexpr int LDT = 48;                 // LDS row stride (halves) incl pad
    __shared__ alignas(16) bf16_t sA[2][128 * LDT];
    __shared__ alignas(16) bf16_t sB[2][128 * LDT];

    const int tid    = threadIdx.x;
    const int lane   = tid & 31;
    const int wid    = tid >> 5;            // 0..7
    const int wave_m = wid & 3;             // 4 row-groups of 32
    const int wave_n = wid >> 2;            // 2 col-groups of 64
    const int n0     = blockIdx.x * 128;
    const int m0     = blockIdx.y * 128;

    const bf16_t* gA = A  + (size_t)m0 * K; // tile row starts; +k0 per step
    const bf16_t* gB = Wt + (size_t)n0 * K;

    v8f acc[2][4];
#pragma unroll
    for (int i = 0; i < 2; ++i)
#pragma unroll
        for (int j = 0; j < 4; ++j)
#pragma unroll
            for (int e = 0; e < 8; ++e) acc[i][j][e] = 0.f;

    if (wid == 0) {                         // prologue: stage first tiles
        tdm_load_tile_2d(lds_off(&sA[0][0]), gA, 32, 128, K, K, M, true);
        tdm_load_tile_2d(lds_off(&sB[0][0]), gB, 32, 128, K, K, N, true);
    }

    int ib = 0;
    for (int k0 = 0; k0 < K; k0 += 32, ib ^= 1) {
        if (wid == 0) __builtin_amdgcn_s_wait_tensorcnt(0);
        __syncthreads();                    // tile ib ready for everyone
        if (wid == 0 && (k0 + 32) < K) {    // overlap DMA of next tile
            tdm_load_tile_2d(lds_off(&sA[ib ^ 1][0]), gA + k0 + 32,
                             32, 128, K, K, M, true);
            tdm_load_tile_2d(lds_off(&sB[ib ^ 1][0]), gB + k0 + 32,
                             32, 128, K, K, N, true);
        }

        v16bf af[2];
#pragma unroll
        for (int mi = 0; mi < 2; ++mi)
            af[mi] = load_frag(&sA[ib][(wave_m * 32 + mi * 16) * LDT], LDT, lane);
#pragma unroll
        for (int ni = 0; ni < 4; ++ni) {
            v16bf bf = load_frag(&sB[ib][(wave_n * 64 + ni * 16) * LDT], LDT, lane);
#pragma unroll
            for (int mi = 0; mi < 2; ++mi)
                acc[mi][ni] = __builtin_amdgcn_wmma_f32_16x16x32_bf16(
                    false, af[mi], false, bf, (short)0, acc[mi][ni],
                    false, false);
        }
    }

    // epilogue: lane layout of C tile: col = lane&15, rows r + 8*(lane>>4)
    const int hi = lane >> 4;
#pragma unroll
    for (int mi = 0; mi < 2; ++mi) {
#pragma unroll
        for (int ni = 0; ni < 4; ++ni) {
            const int col = n0 + wave_n * 64 + ni * 16 + (lane & 15);
            const float bv = bias[col];
#pragma unroll
            for (int r = 0; r < 8; ++r) {
                const int row = m0 + wave_m * 32 + mi * 16 + r + 8 * hi;
                float v = acc[mi][ni][r] + bv;
                size_t o = (size_t)row * N + col;
                if (EPI == 0) {
                    outF[o] = v;
                } else if (EPI == 1) {
                    outF[o] = v + res[o];
                } else {
                    outH[o] = f2bf(0.5f * v * (1.0f + erff(v * 0.70710678118f)));
                }
            }
        }
    }
}

// ---------------------------------------------------------------------------
// Kernel 3: split QKV, apply 3-D RoPE to Q,K; write Q,K [b,h,s,dh] bf16 and
// V transposed [b,h,dh,s] bf16 (so attention P@V B-operand is kv-contiguous).
// One thread per (b,s,h).
// ---------------------------------------------------------------------------
__global__ void k_rope_split(const float* __restrict__ qkv,
                             const float* __restrict__ coords,
                             bf16_t* __restrict__ Qr, bf16_t* __restrict__ Kr,
                             bf16_t* __restrict__ Vt) {
    int idx = blockIdx.x * 256 + threadIdx.x;      // B*S*H = 65536
    if (idx >= Bc * Sc * Hc) return;
    const int h  = idx & (Hc - 1);
    const int bs = idx >> 3;                        // b*S + s
    const int s  = bs & (Sc - 1);
    const int b  = bs >> 11;

    const size_t in_base = (size_t)bs * (3 * Dc);
    const float* q_in = qkv + in_base + 0 * Dc + h * DHc;
    const float* k_in = qkv + in_base + 1 * Dc + h * DHc;
    const float* v_in = qkv + in_base + 2 * Dc + h * DHc;

    const size_t bh  = (size_t)(b * Hc + h);
    bf16_t* q_out = Qr + (bh * Sc + s) * DHc;
    bf16_t* k_out = Kr + (bh * Sc + s) * DHc;

    const float c0 = coords[(size_t)bs * 3 + 0];
    const float c1 = coords[(size_t)bs * 3 + 1];
    const float c2 = coords[(size_t)bs * 3 + 2];
    const float cax[3] = {c0, c1, c2};
    const float lg = -logf(10000.0f) / 16.0f;

#pragma unroll
    for (int ax = 0; ax < 3; ++ax) {
        const float c = cax[ax];
#pragma unroll 4
        for (int j = 0; j < 16; ++j) {
            const float ang = c * __expf(lg * (float)j);
            float cs, sn;
            __sincosf(ang, &sn, &cs);
            const int i1 = ax * 32 + j, i2 = i1 + 16;
            float q1 = q_in[i1], q2 = q_in[i2];
            q_out[i1] = f2bf(q1 * cs - q2 * sn);
            q_out[i2] = f2bf(q1 * sn + q2 * cs);
            float k1 = k_in[i1], k2 = k_in[i2];
            k_out[i1] = f2bf(k1 * cs - k2 * sn);
            k_out[i2] = f2bf(k1 * sn + k2 * cs);
        }
    }
    bf16_t* v_out = Vt + bh * (size_t)(DHc * Sc);
#pragma unroll 8
    for (int d = 0; d < DHc; ++d)
        v_out[(size_t)d * Sc + s] = f2bf(v_in[d]);
}

// ---------------------------------------------------------------------------
// Kernel 4: causal flash attention. Block = 128 thr = 4 waves; each wave owns
// a 16-row query tile; 32-wide KV steps; online softmax; WMMA for both
// Q@K^T (3 k-steps over DH=96) and P@V (6 output dh tiles).
// ---------------------------------------------------------------------------
__global__ __launch_bounds__(128)
void k_flash_attn(const bf16_t* __restrict__ Qr, const bf16_t* __restrict__ Kr,
                  const bf16_t* __restrict__ Vt, bf16_t* __restrict__ Aout) {
    __shared__ alignas(16) bf16_t sK[32 * DHc];     // [kv=32][dh=96]
    __shared__ alignas(16) bf16_t sV[DHc * 32];     // [dh=96][kv=32]
    __shared__ alignas(16) bf16_t sP[4 * 16 * 32];  // per-wave P [q=16][kv=32]

    const int tid  = threadIdx.x;
    const int lane = tid & 31;
    const int wid  = tid >> 5;
    const int qblk = blockIdx.x;          // 0..31  (64 queries per block)
    const int h    = blockIdx.y;
    const int b    = blockIdx.z;
    const size_t bh = (size_t)(b * Hc + h);

    const int q0 = qblk * 64 + wid * 16;  // wave's query tile base
    const float scale = 0.10206207262f;   // 1/sqrt(96)

    // Q fragments (3 k-steps of 32 over DH=96), loaded once from global
    const bf16_t* qbase = Qr + (bh * Sc + q0) * DHc;
    v16bf qf[3];
#pragma unroll
    for (int f = 0; f < 3; ++f) qf[f] = load_frag(qbase + f * 32, DHc, lane);

    v8f o[6];
#pragma unroll
    for (int d = 0; d < 6; ++d)
#pragma unroll
        for (int e = 0; e < 8; ++e) o[d][e] = 0.f;
    float mrow[8], lrow[8];
#pragma unroll
    for (int r = 0; r < 8; ++r) { mrow[r] = -1e30f; lrow[r] = 0.f; }

    const int hi = lane >> 4;
    const int nsteps = qblk * 2 + 2;      // causal: kv <= q_block end
    for (int step = 0; step < nsteps; ++step) {
        const int kv0 = step * 32;
        __syncthreads();
        // stage K [32][96] and V^T [96][32]; 3x b128 each per thread
        {
            const bf16_t* kg = Kr + (bh * Sc + kv0) * DHc;
            const bf16_t* vg = Vt + bh * (size_t)(DHc * Sc) + kv0;
#pragma unroll
            for (int i = 0; i < 3; ++i) {
                int t = tid + i * 128;
                int krow = t / 12, kseg = (t % 12) * 8;
                *(v8bf*)(sK + krow * DHc + kseg) =
                    *(const v8bf*)(kg + (size_t)krow * DHc + kseg);
                int vrow = t >> 2, vseg = (t & 3) * 8;
                *(v8bf*)(sV + vrow * 32 + vseg) =
                    *(const v8bf*)(vg + (size_t)vrow * Sc + vseg);
            }
        }
        __syncthreads();

        // scores: two 16x16 tiles covering kv0..kv0+31
        v8f sc[2];
#pragma unroll
        for (int t2 = 0; t2 < 2; ++t2) {
#pragma unroll
            for (int e = 0; e < 8; ++e) sc[t2][e] = 0.f;
#pragma unroll
            for (int f = 0; f < 3; ++f) {
                v16bf kf = load_frag(sK + (t2 * 16) * DHc + f * 32, DHc, lane);
                sc[t2] = __builtin_amdgcn_wmma_f32_16x16x32_bf16(
                    false, qf[f], false, kf, (short)0, sc[t2], false, false);
            }
        }
        // scale + causal mask
#pragma unroll
        for (int t2 = 0; t2 < 2; ++t2) {
            const int col = kv0 + t2 * 16 + (lane & 15);
#pragma unroll
            for (int r = 0; r < 8; ++r) {
                const int row = q0 + r + 8 * hi;
                float v = sc[t2][r] * scale;
                sc[t2][r] = (col > row) ? -1e30f : v;
            }
        }
        // online softmax (row spread over 16 lanes of the half-wave)
        float al[8];
#pragma unroll
        for (int r = 0; r < 8; ++r) {
            float mx = fmaxf(sc[0][r], sc[1][r]);
            mx = fmaxf(mx, __shfl_xor(mx, 1, 32));
            mx = fmaxf(mx, __shfl_xor(mx, 2, 32));
            mx = fmaxf(mx, __shfl_xor(mx, 4, 32));
            mx = fmaxf(mx, __shfl_xor(mx, 8, 32));
            const float mn = fmaxf(mrow[r], mx);
            al[r] = __expf(mrow[r] - mn);
            mrow[r] = mn;
            sc[0][r] = __expf(sc[0][r] - mn);
            sc[1][r] = __expf(sc[1][r] - mn);
            float s = sc[0][r] + sc[1][r];
            s += __shfl_xor(s, 1, 32);
            s += __shfl_xor(s, 2, 32);
            s += __shfl_xor(s, 4, 32);
            s += __shfl_xor(s, 8, 32);
            lrow[r] = lrow[r] * al[r] + s;
        }
#pragma unroll
        for (int d = 0; d < 6; ++d)
#pragma unroll
            for (int r = 0; r < 8; ++r) o[d][r] *= al[r];

        // stage P (bf16) into this wave's LDS region, reload as A fragment
        bf16_t* pw = sP + wid * (16 * 32);
#pragma unroll
        for (int t2 = 0; t2 < 2; ++t2)
#pragma unroll
            for (int r = 0; r < 8; ++r)
                pw[(r + 8 * hi) * 32 + t2 * 16 + (lane & 15)] = f2bf(sc[t2][r]);
        v16bf pf = load_frag(pw, 32, lane);   // LDS ops in-order per wave

        // O += P @ V  (6 dh tiles, one 32-deep k-step each)
#pragma unroll
        for (int d = 0; d < 6; ++d) {
            v16bf vf = load_frag(sV + (d * 16) * 32, 32, lane);
            o[d] = __builtin_amdgcn_wmma_f32_16x16x32_bf16(
                false, pf, false, vf, (short)0, o[d], false, false);
        }
    }

    // normalize and scatter to [b*s][D] bf16 (heads merged)
    float inv[8];
#pragma unroll
    for (int r = 0; r < 8; ++r) inv[r] = 1.0f / lrow[r];
#pragma unroll
    for (int d = 0; d < 6; ++d) {
        const int col = h * DHc + d * 16 + (lane & 15);
#pragma unroll
        for (int r = 0; r < 8; ++r) {
            const int row = q0 + r + 8 * hi;
            Aout[((size_t)b * Sc + row) * Dc + col] = f2bf(o[d][r] * inv[r]);
        }
    }
}

// ---------------------------------------------------------------------------
// Host-side launch
// ---------------------------------------------------------------------------
extern "C" void kernel_launch(void* const* d_in, const int* in_sizes, int n_in,
                              void* d_out, int out_size, void* d_ws,
                              size_t ws_size, hipStream_t stream) {
    const float* src     = (const float*)d_in[0];
    const float* coords  = (const float*)d_in[1];
    // d_in[2] = causal mask (handled analytically)
    const float* n1_g    = (const float*)d_in[3];
    const float* n1_b    = (const float*)d_in[4];
    const float* qkv_w   = (const float*)d_in[5];
    const float* qkv_b   = (const float*)d_in[6];
    const float* proj_w  = (const float*)d_in[7];
    const float* proj_b  = (const float*)d_in[8];
    const float* n2_g    = (const float*)d_in[9];
    const float* n2_b    = (const float*)d_in[10];
    const float* fc1_w   = (const float*)d_in[11];
    const float* fc1_b   = (const float*)d_in[12];
    const float* fc2_w   = (const float*)d_in[13];
    const float* fc2_b   = (const float*)d_in[14];
    float* out = (float*)d_out;

    // workspace carve-out
    char* ws = (char*)d_ws;
    size_t off = 0;
    auto alloc = [&](size_t bytes) -> char* {
        char* p = ws + off;
        off += (bytes + 255) & ~(size_t)255;
        return p;
    };
    bf16_t* xn      = (bf16_t*)alloc((size_t)Mrows * Dc * 2);
    bf16_t* wqkv_t  = (bf16_t*)alloc((size_t)(3 * Dc) * Dc * 2);
    bf16_t* wproj_t = (bf16_t*)alloc((size_t)Dc * Dc * 2);
    bf16_t* wfc1_t  = (bf16_t*)alloc((size_t)HIDc * Dc * 2);
    bf16_t* wfc2_t  = (bf16_t*)alloc((size_t)Dc * HIDc * 2);
    float*  qkv     = (float*) alloc((size_t)Mrows * (3 * Dc) * 4);
    bf16_t* Qr      = (bf16_t*)alloc((size_t)Mrows * Dc * 2);
    bf16_t* Kr      = (bf16_t*)alloc((size_t)Mrows * Dc * 2);
    bf16_t* Vt      = (bf16_t*)alloc((size_t)Mrows * Dc * 2);
    bf16_t* attn    = (bf16_t*)alloc((size_t)Mrows * Dc * 2);
    float*  xres    = (float*) alloc((size_t)Mrows * Dc * 4);
    bf16_t* ybuf    = (bf16_t*)alloc((size_t)Mrows * Dc * 2);
    bf16_t* hbuf    = (bf16_t*)alloc((size_t)Mrows * HIDc * 2);
    (void)ws_size; (void)in_sizes; (void)n_in; (void)out_size;

    // weights -> bf16 transposed
    auto cvt = [&](const float* W, bf16_t* Wt, int K, int N) {
        int n = N * K;
        k_convert_transpose<<<(n + 255) / 256, 256, 0, stream>>>(W, Wt, K, N);
    };
    cvt(qkv_w,  wqkv_t,  Dc,   3 * Dc);
    cvt(proj_w, wproj_t, Dc,   Dc);
    cvt(fc1_w,  wfc1_t,  Dc,   HIDc);
    cvt(fc2_w,  wfc2_t,  HIDc, Dc);

    // LN1
    k_layernorm_bf16<<<Mrows, 256, 0, stream>>>(src, n1_g, n1_b, xn);

    // QKV GEMM: [8192,768] x [768,2304] -> fp32
    k_gemm_bf16<0><<<dim3((3 * Dc) / 128, Mrows / 128), 256, 0, stream>>>(
        xn, wqkv_t, qkv_b, nullptr, qkv, nullptr, Mrows, 3 * Dc, Dc);

    // RoPE split
    k_rope_split<<<(Bc * Sc * Hc) / 256, 256, 0, stream>>>(qkv, coords, Qr, Kr, Vt);

    // causal flash attention
    k_flash_attn<<<dim3(Sc / 64, Hc, Bc), 128, 0, stream>>>(Qr, Kr, Vt, attn);

    // proj GEMM + src residual -> x (fp32)
    k_gemm_bf16<1><<<dim3(Dc / 128, Mrows / 128), 256, 0, stream>>>(
        attn, wproj_t, proj_b, src, xres, nullptr, Mrows, Dc, Dc);

    // LN2
    k_layernorm_bf16<<<Mrows, 256, 0, stream>>>(xres, n2_g, n2_b, ybuf);

    // fc1 GEMM + GELU -> bf16
    k_gemm_bf16<2><<<dim3(HIDc / 128, Mrows / 128), 256, 0, stream>>>(
        ybuf, wfc1_t, fc1_b, nullptr, nullptr, hbuf, Mrows, HIDc, Dc);

    // fc2 GEMM + x residual -> out (fp32)
    k_gemm_bf16<1><<<dim3(Dc / 128, Mrows / 128), 256, 0, stream>>>(
        hbuf, wfc2_t, fc2_b, xres, out, nullptr, Mrows, Dc, HIDc);
}
